// RoPEMultiHeadAttention_36206574306045
// MI455X (gfx1250) — matrix operand order repaired and checked by reading, not compiled, not run on hardware
//
#include <hip/hip_runtime.h>

// ---------------------------------------------------------------------------
// RoPE multi-head attention for MI455X (gfx1250, wave32, WMMA bf16 16x16x32)
// B=4, S=2048, D=1024, H=16, HD=64.  fp32 in/out, bf16 tensor-core compute.
// v4: manual ping-pong (2-stage) GEMM pipeline -> no inter-stage reg copies.
// ---------------------------------------------------------------------------

typedef __attribute__((ext_vector_type(16))) __bf16 v16bf;
typedef __attribute__((ext_vector_type(8)))  float  v8f;
typedef int v4i_async __attribute__((vector_size(16)));

#define S_LEN  2048
#define DMODEL 1024
#define NHEAD  16
#define HDIM   64
#define NB     4

#ifndef __has_builtin
#define __has_builtin(x) 0
#endif
#if __has_builtin(__builtin_amdgcn_global_load_async_to_lds_b128)
#define HAVE_ASYNC_LDS 1
#else
#define HAVE_ASYNC_LDS 0
#endif

__device__ __forceinline__ unsigned short f2bf(float f) {
    unsigned int u = __float_as_uint(f);
    u += 0x7fffu + ((u >> 16) & 1u);          // round-to-nearest-even
    return (unsigned short)(u >> 16);
}

union Frag16 {          // 16 bf16 = 8 VGPRs, loaded as two b128
    v16bf v;
    uint4 q[2];
};

__device__ __forceinline__ v8f wmma_bf16(const Frag16& a, const Frag16& b, v8f c) {
    return __builtin_amdgcn_wmma_f32_16x16x32_bf16(
        false, a.v, false, b.v, (short)0, c, false, false);
}

// 16-byte async global->LDS copy (falls back to load+ds_store if unavailable).
__device__ __forceinline__ void async_cp16(unsigned short* dst_lds,
                                           const unsigned short* src_g) {
#if HAVE_ASYNC_LDS
    __builtin_amdgcn_global_load_async_to_lds_b128(
        (v4i_async*)const_cast<unsigned short*>(src_g),
        (v4i_async*)dst_lds,
        0, 0);
#else
    *reinterpret_cast<uint4*>(dst_lds) = *reinterpret_cast<const uint4*>(src_g);
#endif
}

__device__ __forceinline__ void async_wait_le4() {
#if HAVE_ASYNC_LDS
#if __has_builtin(__builtin_amdgcn_s_wait_asynccnt)
    __builtin_amdgcn_s_wait_asynccnt(4);
#else
    asm volatile("s_wait_asynccnt 4" ::: "memory");
#endif
#endif
}
__device__ __forceinline__ void async_wait_le0() {
#if HAVE_ASYNC_LDS
#if __has_builtin(__builtin_amdgcn_s_wait_asynccnt)
    __builtin_amdgcn_s_wait_asynccnt(0);
#else
    asm volatile("s_wait_asynccnt 0" ::: "memory");
#endif
#endif
}

// helpers for fragment loads
__device__ __forceinline__ void load_a(Frag16& f, const unsigned short* p) {
    f.q[0] = *reinterpret_cast<const uint4*>(p);
    f.q[1] = *reinterpret_cast<const uint4*>(p + 8);
}
__device__ __forceinline__ void load_b(Frag16& f, const unsigned short* p) {
    f.q[0] = *reinterpret_cast<const uint4*>(p);
    f.q[1] = *reinterpret_cast<const uint4*>(p + 8);
}

// ---------------------------------------------------------------------------
// Kernel 0a: convert + transpose 4 weight matrices f32[k][n] -> bf16 WT[n][k]
// ---------------------------------------------------------------------------
__global__ void cvt_weights(const float* __restrict__ W0, const float* __restrict__ W1,
                            const float* __restrict__ W2, const float* __restrict__ W3,
                            unsigned short* __restrict__ dst) {
    size_t id = (size_t)blockIdx.x * blockDim.x + threadIdx.x;   // 4 * 2^20
    int w = (int)(id >> 20);
    size_t rem = id & ((1u << 20) - 1);
    int k = (int)(rem >> 10);
    int n = (int)(rem & 1023);
    const float* src = (w == 0) ? W0 : (w == 1) ? W1 : (w == 2) ? W2 : W3;
    dst[((size_t)w << 20) + ((size_t)n << 10) + k] = f2bf(src[((size_t)k << 10) + n]);
}

// Kernel 0b: x f32 -> bf16
__global__ void cvt_x(const float* __restrict__ src, unsigned short* __restrict__ dst, int n) {
    int i = blockIdx.x * blockDim.x + threadIdx.x;
    if (i < n) dst[i] = f2bf(src[i]);
}

// ---------------------------------------------------------------------------
// Kernel 1: fused QKV projection + bias + RoPE, ping-pong pipelined.
// One wave = 16 rows x one head (64 cols) of one of {Q,K,V}.
// ---------------------------------------------------------------------------
__global__ void __launch_bounds__(256)
qkv_rope(const unsigned short* __restrict__ xb,
         const unsigned short* __restrict__ Wt,      // [3][1024][1024] bf16 (n-major)
         const float* __restrict__ bq, const float* __restrict__ bk,
         const float* __restrict__ bv,
         unsigned short* __restrict__ Qr, unsigned short* __restrict__ Kr,
         unsigned short* __restrict__ Vt) {
    const int wid  = blockIdx.x * 8 + (threadIdx.x >> 5);
    const int lane = threadIdx.x & 31;
    const int h    = lane >> 4;
    const int n16  = lane & 15;

    const int mat  = wid / (512 * NHEAD);            // 0=Q 1=K 2=V
    const int rem  = wid - mat * 512 * NHEAD;
    const int head = rem >> 9;
    const int rt   = rem & 511;
    const int colbase = head * HDIM;

    const unsigned short* Wsel = Wt + ((size_t)mat << 20);
    const float* bias = (mat == 0) ? bq : (mat == 1) ? bk : bv;

    v8f acc[4];
    const v8f vzero = {0.f, 0.f, 0.f, 0.f, 0.f, 0.f, 0.f, 0.f};
#pragma unroll
    for (int nt = 0; nt < 4; ++nt) acc[nt] = vzero;

    const unsigned short* arow = xb + (((size_t)rt * 16 + n16) << 10) + h * 16;
    const unsigned short* brow[4];
#pragma unroll
    for (int nt = 0; nt < 4; ++nt)
        brow[nt] = Wsel + (((size_t)(colbase + nt * 16 + n16)) << 10) + h * 16;

    // --- 2-stage ping-pong: WMMAs on one buffer set while loads fill other --
    Frag16 a0, a1, b0[4], b1[4];
    load_a(a0, arow);
#pragma unroll
    for (int nt = 0; nt < 4; ++nt) load_b(b0[nt], brow[nt]);

    for (int kk = 0; kk < DMODEL; kk += 64) {
        load_a(a1, arow + kk + 32);                     // odd stage (always valid)
#pragma unroll
        for (int nt = 0; nt < 4; ++nt) load_b(b1[nt], brow[nt] + kk + 32);
#pragma unroll
        for (int nt = 0; nt < 4; ++nt) acc[nt] = wmma_bf16(a0, b0[nt], acc[nt]);

        if (kk + 64 < DMODEL) {                         // even stage prefetch
            load_a(a0, arow + kk + 64);
#pragma unroll
            for (int nt = 0; nt < 4; ++nt) load_b(b0[nt], brow[nt] + kk + 64);
        }
#pragma unroll
        for (int nt = 0; nt < 4; ++nt) acc[nt] = wmma_bf16(a1, b1[nt], acc[nt]);
    }

    // bias (zeros here, kept for fidelity)
#pragma unroll
    for (int nt = 0; nt < 4; ++nt) {
        float bn = bias[colbase + nt * 16 + n16];
#pragma unroll
        for (int e = 0; e < 8; ++e) acc[nt][e] += bn;
    }

    // RoPE: pairs (d, d+32) live in fragment pairs (nt, nt+2), same lane/elem.
    if (mat < 2) {
#pragma unroll
        for (int ntl = 0; ntl < 2; ++ntl) {
            const int d_lo = ntl * 16 + n16;
            const float inv = __powf(10000.0f, -(float)d_lo * (1.0f / 32.0f));
#pragma unroll
            for (int e = 0; e < 8; ++e) {
                const int row = rt * 16 + e + 8 * h;
                const float t = (float)(row & (S_LEN - 1));
                float sn, cs;
                __sincosf(t * inv, &sn, &cs);
                const float qlo = acc[ntl][e], qhi = acc[ntl + 2][e];
                acc[ntl][e]     = qlo * cs - qhi * sn;
                acc[ntl + 2][e] = qhi * cs + qlo * sn;
            }
        }
    }

#pragma unroll
    for (int nt = 0; nt < 4; ++nt) {
#pragma unroll
        for (int e = 0; e < 8; ++e) {
            const int m = e + 8 * h;
            const int row = rt * 16 + m;
            const int b = row >> 11, s = row & (S_LEN - 1);
            const int d = nt * 16 + n16;
            const unsigned short val = f2bf(acc[nt][e]);
            if (mat == 0)
                Qr[(((size_t)(b * NHEAD + head) * S_LEN + s) << 6) + d] = val;
            else if (mat == 1)
                Kr[(((size_t)(b * NHEAD + head) * S_LEN + s) << 6) + d] = val;
            else
                Vt[(((size_t)((b * NHEAD + head) * HDIM + d)) << 11) + s] = val;
        }
    }
}

// ---------------------------------------------------------------------------
// Kernel 2: flash attention with double-buffered ASYNC-to-LDS K/V staging.
// Block = 8 waves sharing one (b,h); each wave owns a 16-query tile.
// ---------------------------------------------------------------------------
__global__ void __launch_bounds__(256)
flash_attn(const unsigned short* __restrict__ Qr, const unsigned short* __restrict__ Kr,
           const unsigned short* __restrict__ Vt, unsigned short* __restrict__ attn) {
    __shared__ unsigned short kbuf[2][64 * HDIM];    // [kv][d]   2 x 8 KB
    __shared__ unsigned short vbuf[2][HDIM * 64];    // [d][kv]   2 x 8 KB
    __shared__ unsigned short pbuf[8 * 16 * HDIM];   // per-wave P tiles, 16 KB

    const int tid  = threadIdx.x;
    const int w    = tid >> 5;
    const int lane = tid & 31;
    const int h    = lane >> 4;
    const int n16  = lane & 15;

    const int bh = blockIdx.x >> 4;                  // 0..63  (b*16+head)
    const int qt = ((blockIdx.x & 15) << 3) + w;     // 0..127
    const int b  = bh >> 4, head = bh & 15;
    const size_t rowbase = (size_t)bh * S_LEN;

    const unsigned short* Kbase = Kr + (rowbase << 6);       // [2048][64]
    const unsigned short* Vbase = Vt + ((size_t)bh << 17);   // [64][2048]

    // Q A-fragments, resident for the whole loop
    Frag16 qa0, qa1;
    {
        const unsigned short* qrow = Qr + ((rowbase + qt * 16 + n16) << 6);
        load_a(qa0, qrow + h * 16);
        load_a(qa1, qrow + 32 + h * 16);
    }

    // tile staging: K tile is 4096 contiguous bf16; V tile is 64 rows x 128B
    auto issue_tiles = [&](int kb, int buf) {
#pragma unroll
        for (int p = 0; p < 2; ++p) {                        // K: 2 x 16B/thread
            const int off = tid * 8 + p * 2048;
            async_cp16(&kbuf[buf][off], Kbase + ((size_t)kb << 6) + off);
        }
#pragma unroll
        for (int p = 0; p < 2; ++p) {                        // V: 2 x 16B/thread
            const int r = (tid >> 3) + p * 32;
            const int c = (tid & 7) * 8;
            async_cp16(&vbuf[buf][r * 64 + c], Vbase + (size_t)r * S_LEN + kb + c);
        }
    };

    const v8f vzero = {0.f, 0.f, 0.f, 0.f, 0.f, 0.f, 0.f, 0.f};
    v8f o[4];
    float row_m[8], row_l[8];
#pragma unroll
    for (int nt = 0; nt < 4; ++nt) o[nt] = vzero;
#pragma unroll
    for (int e = 0; e < 8; ++e) { row_m[e] = -3.0e38f; row_l[e] = 0.f; }

    unsigned short* pb = pbuf + w * 16 * HDIM;

    issue_tiles(0, 0);                                       // prologue

    for (int kb = 0; kb < S_LEN; kb += 64) {
        const int p = (kb >> 6) & 1;
        const int kn = kb + 64;
        if (kn < S_LEN) {
            issue_tiles(kn, 1 - p);                          // DMA next tile
            async_wait_le4();                                // current tile done
        } else {
            async_wait_le0();
        }
        __syncthreads();                                     // tile visible to all

        // ----- S = Q K^T (K B-fragments from LDS) -----
        v8f sf[4];
#pragma unroll
        for (int nt = 0; nt < 4; ++nt) {
            const unsigned short* krow = &kbuf[p][(nt * 16 + n16) * HDIM];
            Frag16 kb0, kb1;
            load_b(kb0, krow + h * 16);
            load_b(kb1, krow + 32 + h * 16);
            v8f c = vzero;
            c = wmma_bf16(qa0, kb0, c);
            c = wmma_bf16(qa1, kb1, c);
            sf[nt] = c;
        }

#pragma unroll
        for (int nt = 0; nt < 4; ++nt)
#pragma unroll
            for (int e = 0; e < 8; ++e) sf[nt][e] *= 0.125f;   // 1/sqrt(64)

        // ----- online softmax (rows across 16 lanes of a half-wave) -----
        float fac[8], mc[8], ls[8];
#pragma unroll
        for (int e = 0; e < 8; ++e) {
            float v = fmaxf(fmaxf(sf[0][e], sf[1][e]), fmaxf(sf[2][e], sf[3][e]));
            v = fmaxf(v, __shfl_xor(v, 1, 32));
            v = fmaxf(v, __shfl_xor(v, 2, 32));
            v = fmaxf(v, __shfl_xor(v, 4, 32));
            v = fmaxf(v, __shfl_xor(v, 8, 32));
            mc[e]    = fmaxf(row_m[e], v);
            fac[e]   = __expf(row_m[e] - mc[e]);
            row_m[e] = mc[e];
            ls[e]    = 0.f;
        }
#pragma unroll
        for (int nt = 0; nt < 4; ++nt)
#pragma unroll
            for (int e = 0; e < 8; ++e) {
                const float pp = __expf(sf[nt][e] - mc[e]);
                sf[nt][e] = pp;
                ls[e] += pp;
            }
#pragma unroll
        for (int e = 0; e < 8; ++e) {
            float v = ls[e];
            v += __shfl_xor(v, 1, 32);
            v += __shfl_xor(v, 2, 32);
            v += __shfl_xor(v, 4, 32);
            v += __shfl_xor(v, 8, 32);
            row_l[e] = row_l[e] * fac[e] + v;
        }
#pragma unroll
        for (int nt = 0; nt < 4; ++nt)
#pragma unroll
            for (int e = 0; e < 8; ++e) o[nt][e] *= fac[e];

        // ----- P: C-fragment -> per-wave LDS slice -> A-fragment -----
#pragma unroll
        for (int nt = 0; nt < 4; ++nt)
#pragma unroll
            for (int e = 0; e < 8; ++e)
                pb[(e + 8 * h) * HDIM + nt * 16 + n16] = f2bf(sf[nt][e]);
        asm volatile("s_wait_dscnt 0" ::: "memory");   // same-wave LDS RAW

        Frag16 pa0, pa1;
        {
            const unsigned short* prow = pb + n16 * HDIM;
            load_a(pa0, prow + h * 16);
            load_a(pa1, prow + 32 + h * 16);
        }

        // ----- O += P V  (V B-fragments from LDS, transposed layout) -----
#pragma unroll
        for (int nt = 0; nt < 4; ++nt) {
            const unsigned short* vrow = &vbuf[p][(nt * 16 + n16) * 64];
            Frag16 vb0, vb1;
            load_b(vb0, vrow + h * 16);
            load_b(vb1, vrow + 32 + h * 16);
            o[nt] = wmma_bf16(pa0, vb0, o[nt]);
            o[nt] = wmma_bf16(pa1, vb1, o[nt]);
        }

        __syncthreads();   // all waves done with buf[p] before it is refilled
    }

    float inv[8];
#pragma unroll
    for (int e = 0; e < 8; ++e) inv[e] = 1.0f / row_l[e];
#pragma unroll
    for (int nt = 0; nt < 4; ++nt)
#pragma unroll
        for (int e = 0; e < 8; ++e) {
            const int srow = qt * 16 + e + 8 * h;
            attn[(((size_t)b * S_LEN + srow) << 10) + head * HDIM + nt * 16 + n16] =
                f2bf(o[nt][e] * inv[e]);
        }
}

// ---------------------------------------------------------------------------
// Kernel 3: output projection, ping-pong pipelined GEMM -> f32 out
// ---------------------------------------------------------------------------
__global__ void __launch_bounds__(256)
out_proj(const unsigned short* __restrict__ attn, const unsigned short* __restrict__ WoT,
         const float* __restrict__ bo, float* __restrict__ out) {
    const int wid  = blockIdx.x * 8 + (threadIdx.x >> 5);
    const int lane = threadIdx.x & 31;
    const int h    = lane >> 4;
    const int n16  = lane & 15;
    const int strip = wid >> 9;
    const int rt    = wid & 511;
    const int colbase = strip * 64;

    const v8f vzero = {0.f, 0.f, 0.f, 0.f, 0.f, 0.f, 0.f, 0.f};
    v8f acc[4];
#pragma unroll
    for (int nt = 0; nt < 4; ++nt) acc[nt] = vzero;

    const unsigned short* arow = attn + (((size_t)rt * 16 + n16) << 10) + h * 16;
    const unsigned short* brow[4];
#pragma unroll
    for (int nt = 0; nt < 4; ++nt)
        brow[nt] = WoT + (((size_t)(colbase + nt * 16 + n16)) << 10) + h * 16;

    Frag16 a0, a1, b0[4], b1[4];
    load_a(a0, arow);
#pragma unroll
    for (int nt = 0; nt < 4; ++nt) load_b(b0[nt], brow[nt]);

    for (int kk = 0; kk < DMODEL; kk += 64) {
        load_a(a1, arow + kk + 32);
#pragma unroll
        for (int nt = 0; nt < 4; ++nt) load_b(b1[nt], brow[nt] + kk + 32);
#pragma unroll
        for (int nt = 0; nt < 4; ++nt) acc[nt] = wmma_bf16(a0, b0[nt], acc[nt]);

        if (kk + 64 < DMODEL) {
            load_a(a0, arow + kk + 64);
#pragma unroll
            for (int nt = 0; nt < 4; ++nt) load_b(b0[nt], brow[nt] + kk + 64);
        }
#pragma unroll
        for (int nt = 0; nt < 4; ++nt) acc[nt] = wmma_bf16(a1, b1[nt], acc[nt]);
    }

#pragma unroll
    for (int nt = 0; nt < 4; ++nt) {
        const float bn = bo[colbase + nt * 16 + n16];
#pragma unroll
        for (int e = 0; e < 8; ++e) {
            const int row = rt * 16 + e + 8 * h;
            out[((size_t)row << 10) + colbase + nt * 16 + n16] = acc[nt][e] + bn;
        }
    }
}

// ---------------------------------------------------------------------------
extern "C" void kernel_launch(void* const* d_in, const int* in_sizes, int n_in,
                              void* d_out, int out_size, void* d_ws, size_t ws_size,
                              hipStream_t stream) {
    const float* x  = (const float*)d_in[0];
    const float* Wq = (const float*)d_in[1];
    const float* bq = (const float*)d_in[2];
    const float* Wk = (const float*)d_in[3];
    const float* bk = (const float*)d_in[4];
    const float* Wv = (const float*)d_in[5];
    const float* bv = (const float*)d_in[6];
    const float* Wo = (const float*)d_in[7];
    const float* bo = (const float*)d_in[8];

    const size_t N_ROWS = (size_t)NB * S_LEN;          // 8192
    char* ws = (char*)d_ws;
    unsigned short* xb   = (unsigned short*)ws;                         // 16 MB
    unsigned short* Wt   = (unsigned short*)(ws + N_ROWS * DMODEL * 2); //  8 MB
    unsigned short* Qr   = (unsigned short*)((char*)Wt + (size_t)4 * DMODEL * DMODEL * 2);
    unsigned short* Kr   = (unsigned short*)((char*)Qr + N_ROWS * DMODEL * 2);
    unsigned short* Vt   = (unsigned short*)((char*)Kr + N_ROWS * DMODEL * 2);
    unsigned short* attn = (unsigned short*)((char*)Vt + N_ROWS * DMODEL * 2);

    cvt_weights<<<(4u << 20) / 256, 256, 0, stream>>>(Wq, Wk, Wv, Wo, Wt);
    const int nx = NB * S_LEN * DMODEL;
    cvt_x<<<(nx + 255) / 256, 256, 0, stream>>>(x, xb, nx);

    qkv_rope<<<3072, 256, 0, stream>>>(xb, Wt, bq, bk, bv, Qr, Kr, Vt);
    flash_attn<<<1024, 256, 0, stream>>>(Qr, Kr, Vt, attn);
    out_proj<<<1024, 256, 0, stream>>>(attn, Wt + ((size_t)3 << 20), bo, (float*)d_out);
}
